// GAE_53008486367982
// MI455X (gfx1250) — matrix operand order repaired and checked by reading, not compiled, not run on hardware
//
#include <hip/hip_runtime.h>
#include <hip/hip_bf16.h>

typedef __attribute__((ext_vector_type(16))) __bf16 v16bf;
typedef __attribute__((ext_vector_type(8)))  float  v8f;

#define N_NODES 100000
#define N_EDGES 1600000
#define N_PAIRS 500000
#define NHID    128

// ---------------------------------------------------------------------------
// TDM: DMA a flat 64KB block (8192 x 8B elements) from global into LDS.
// D# built per CDNA5 ISA §8.3-8.5 (group0: count/lds/global/type=2;
// group1: data_size=8B, tensor_dim0=tile_dim0=8192, tile_dim1=1, stride=8192;
// groups 2/3 zero for <=2D). This toolchain exposes the 6-arg builtin form:
// (uint32x4 g0, int32x8 g1, int32x4 g2, int32x4 g3, int32x8 gx, i32 cpol);
// the extra int32x8 is passed zero-filled. Issue from ONE wave only (EXEC is
// ignored by TDM); caller must __syncthreads() after (s_wait_tensorcnt 0 here).
// ---------------------------------------------------------------------------
#if __has_builtin(__builtin_amdgcn_tensor_load_to_lds)
#define HAVE_TDM 1
#endif

__device__ __forceinline__ void stage_64kb(const void* gsrc, void* lds_dst,
                                           int tid, int wave) {
#ifdef HAVE_TDM
  if (wave == 0) {
    typedef __attribute__((ext_vector_type(4))) unsigned v4u_;
    typedef __attribute__((ext_vector_type(8))) int      v8i_;
    typedef __attribute__((ext_vector_type(4))) int      v4i_;
    unsigned long long ga = (unsigned long long)(uintptr_t)gsrc;
    unsigned ldsOff = (unsigned)(uintptr_t)lds_dst;
    v4u_ g0 = { 1u,                                  // count=1, flags=0
                ldsOff,                              // lds_addr
                (unsigned)ga,                        // global_addr[31:0]
                (unsigned)((ga >> 32) & 0x01FFFFFFu) | (2u << 30) }; // addr[56:32]|type=2
    v8i_ g1 = { 0x00030000,   // data_size=3 (8B), mask=0
                0x20000000,   // tensor_dim0[15:0]=8192 @ bits[63:48]
                0x00010000,   // tensor_dim1[15:0]=1    @ bits[95:80]
                0x20000000,   // tile_dim0=8192         @ bits[127:112]
                0x00000001,   // tile_dim1=1            @ bits[143:128]
                0x00002000,   // tensor_dim0_stride=8192
                0, 0 };
    v4i_ gz4 = { 0, 0, 0, 0 };
    v8i_ gz8 = { 0, 0, 0, 0, 0, 0, 0, 0 };
    __builtin_amdgcn_tensor_load_to_lds(g0, g1, gz4, gz4, gz8, 0);
    __builtin_amdgcn_s_wait_tensorcnt(0);
  }
#else
  // Fallback: cooperative copy by the whole block (8192 x 8B)
  unsigned long long* d = (unsigned long long*)lds_dst;
  const unsigned long long* s = (const unsigned long long*)gsrc;
  for (int i = tid; i < 8192; i += 256) d[i] = s[i];
#endif
}

// ---------------------------------------------------------------------------
// Pre-pack weight matrices into WMMA B-fragment layout (bf16).
// B-frag (K=32 x N=16, 16-bit): lane l -> n = l%16, kh = l/16,
//   element e (0..15): K = kstep*32 + kh*16 + e, value = W[K*128 + ntile*16+n]
// Frag f at frag + f*512 bf16 (lane-contiguous 32B). 0..191: 6 GCN mats
// (layer pairs ws||wn are 64KB-contiguous); 192..255: dec_w1 (nt*8+ks).
// ---------------------------------------------------------------------------
__global__ void prep_frags(const float* __restrict__ w0, const float* __restrict__ w1,
                           const float* __restrict__ w2, const float* __restrict__ w3,
                           const float* __restrict__ w4, const float* __restrict__ w5,
                           const float* __restrict__ wdec, __bf16* __restrict__ frag) {
  int g = blockIdx.x * blockDim.x + threadIdx.x;
  int f = g >> 5, lane = g & 31;
  if (f >= 256) return;
  int n = lane & 15, kh = lane >> 4;
  const float* W;
  int nt, ks;
  if (f < 192) {
    int mat = f >> 5, rem = f & 31;
    nt = rem >> 2; ks = rem & 3;
    W = w0;
    if (mat == 1) W = w1;
    if (mat == 2) W = w2;
    if (mat == 3) W = w3;
    if (mat == 4) W = w4;
    if (mat == 5) W = w5;
  } else {
    int r = f - 192; nt = r >> 3; ks = r & 7; W = wdec;
  }
  __bf16* dst = frag + (size_t)(f * 32 + lane) * 16;
#pragma unroll
  for (int e = 0; e < 16; ++e) {
    int K = ks * 32 + kh * 16 + e;
    dst[e] = (__bf16)W[(size_t)K * 128 + nt * 16 + n];
  }
}

// ---------------------------------------------------------------------------
// Dual GEMM: acc = act(h)@Ws + b ; sup = act(h)@Wn. One wave = 16 rows.
// Both weight-frag sets (ws||wn = 64KB) are TDM-staged into LDS once per
// block and shared by all 8 waves. A-frag (16x32 bf16): lane l -> m=l%16,
// kh=l/16; elems 0..15: K=s*32+kh*16+e.
// C/D (16x16 f32): lane l -> n=l%16, mh=l/16; float j -> row mh*8+j.
// In-place (h == accOut) is safe: rows are wave-private, loads before stores.
// ---------------------------------------------------------------------------
__global__ void gemm_dual(const float* __restrict__ h,
                          const __bf16* __restrict__ fragPair,  // ws||wn, 64KB
                          const float* __restrict__ bias,
                          float* __restrict__ accOut,
                          float* __restrict__ supOut,
                          int applyRelu) {
  __shared__ __align__(32) __bf16 sw[32768];            // 64KB
  int tid = threadIdx.x;
  int wave = tid >> 5, lane = tid & 31;

  stage_64kb(fragPair, sw, tid, wave);                   // async DMA (wave 0)

  int rowBase = (blockIdx.x * 8 + wave) * 16;
  bool active = rowBase < N_NODES;                       // wave-uniform
  int m = lane & 15, kh = lane >> 4;

  v16bf A[4];
  if (active) {                                          // overlaps with TDM
    const float* rp = h + (size_t)(rowBase + m) * NHID;
#pragma unroll
    for (int s = 0; s < 4; ++s) {
      int k0 = s * 32 + kh * 16;
      float4 p0 = *(const float4*)(rp + k0);
      float4 p1 = *(const float4*)(rp + k0 + 4);
      float4 p2 = *(const float4*)(rp + k0 + 8);
      float4 p3 = *(const float4*)(rp + k0 + 12);
      float v[16] = {p0.x,p0.y,p0.z,p0.w, p1.x,p1.y,p1.z,p1.w,
                     p2.x,p2.y,p2.z,p2.w, p3.x,p3.y,p3.z,p3.w};
      v16bf a;
#pragma unroll
      for (int e = 0; e < 16; ++e) {
        float x = v[e];
        if (applyRelu) x = fmaxf(x, 0.0f);
        a[e] = (__bf16)x;
      }
      A[s] = a;
    }
  }

  __syncthreads();                                       // LDS weights ready
  if (!active) return;

#pragma unroll
  for (int nt = 0; nt < 8; ++nt) {
    float b = bias[nt * 16 + m];
    v8f c1, c2;
#pragma unroll
    for (int j = 0; j < 8; ++j) { c1[j] = b; c2[j] = 0.0f; }
#pragma unroll
    for (int s = 0; s < 4; ++s) {
      int fo = ((nt * 4 + s) * 32 + lane) * 16;
      v16bf bs = *(const v16bf*)(sw + fo);               // ws frags
      v16bf bn = *(const v16bf*)(sw + 16384 + fo);       // wn frags
      c1 = __builtin_amdgcn_wmma_f32_16x16x32_bf16(false, A[s], false, bs,
                                                   (short)0, c1, false, false);
      c2 = __builtin_amdgcn_wmma_f32_16x16x32_bf16(false, A[s], false, bn,
                                                   (short)0, c2, false, false);
    }
    int n = nt * 16 + m;
#pragma unroll
    for (int j = 0; j < 8; ++j) {
      size_t row = (size_t)(rowBase + kh * 8 + j);
      accOut[row * NHID + n] = c1[j];
      supOut[row * NHID + n] = c2[j];
    }
  }
}

// ---------------------------------------------------------------------------
// Edge scatter: acc[row] += support[col] * w  (one wave per edge, 4 feats/lane)
// support and acc (51.2 MB each) fit in the 192 MB L2 -> atomics resolve in L2.
// ---------------------------------------------------------------------------
__global__ void edge_scatter(const int* __restrict__ erow,
                             const int* __restrict__ ecol,
                             const float* __restrict__ ew,
                             const float* __restrict__ S,
                             float* __restrict__ acc) {
  int e    = (blockIdx.x * blockDim.x + threadIdx.x) >> 5;
  int lane = threadIdx.x & 31;
  if (e >= N_EDGES) return;
  int r = erow[e], c = ecol[e];
  float w = ew[e];
  float4 s = *(const float4*)(S + (size_t)c * NHID + lane * 4);
  float* dst = acc + (size_t)r * NHID + lane * 4;
  __hip_atomic_fetch_add(dst + 0, s.x * w, __ATOMIC_RELAXED, __HIP_MEMORY_SCOPE_AGENT);
  __hip_atomic_fetch_add(dst + 1, s.y * w, __ATOMIC_RELAXED, __HIP_MEMORY_SCOPE_AGENT);
  __hip_atomic_fetch_add(dst + 2, s.z * w, __ATOMIC_RELAXED, __HIP_MEMORY_SCOPE_AGENT);
  __hip_atomic_fetch_add(dst + 3, s.w * w, __ATOMIC_RELAXED, __HIP_MEMORY_SCOPE_AGENT);
}

// ---------------------------------------------------------------------------
// Decoder: pair = [z[xi], z[yi]] (K=256); hdec = relu(pair@W1 + b1);
// p = sigmoid(hdec@w2 + b2). One wave = 16 pairs. dec_w1 frags (64KB) are
// TDM-staged into LDS. The 128->1 dot is done from the D fragments with
// shfl_xor butterfly reductions (no LDS round-trip for hdec).
// ---------------------------------------------------------------------------
__global__ void decoder(const float* __restrict__ z,
                        const int* __restrict__ xi,
                        const int* __restrict__ yi,
                        const __bf16* __restrict__ fragW1,   // 64KB
                        const float* __restrict__ b1,
                        const float* __restrict__ w2,
                        const float* __restrict__ b2,
                        float* __restrict__ pout) {
  __shared__ __align__(32) __bf16 sw[32768];            // 64KB
  int tid = threadIdx.x;
  int wave = tid >> 5, lane = tid & 31;

  stage_64kb(fragW1, sw, tid, wave);                     // async DMA (wave 0)

  int pBase = (blockIdx.x * 8 + wave) * 16;
  bool active = pBase < N_PAIRS;                         // wave-uniform
  int m = lane & 15, kh = lane >> 4;

  v16bf A[8];
  if (active) {
    int p = pBase + m;
    const float* rx = z + (size_t)xi[p] * NHID;
    const float* ry = z + (size_t)yi[p] * NHID;
#pragma unroll
    for (int s = 0; s < 8; ++s) {
      const float* rp = (s < 4) ? rx : ry;
      int k0 = (s & 3) * 32 + kh * 16;
      float4 p0 = *(const float4*)(rp + k0);
      float4 p1 = *(const float4*)(rp + k0 + 4);
      float4 p2 = *(const float4*)(rp + k0 + 8);
      float4 p3 = *(const float4*)(rp + k0 + 12);
      float v[16] = {p0.x,p0.y,p0.z,p0.w, p1.x,p1.y,p1.z,p1.w,
                     p2.x,p2.y,p2.z,p2.w, p3.x,p3.y,p3.z,p3.w};
      v16bf a;
#pragma unroll
      for (int e = 0; e < 16; ++e) a[e] = (__bf16)v[e];
      A[s] = a;
    }
  }

  __syncthreads();
  if (!active) return;

  float accd[8];
#pragma unroll
  for (int j = 0; j < 8; ++j) accd[j] = 0.0f;

#pragma unroll
  for (int nt = 0; nt < 8; ++nt) {
    float bb = b1[nt * 16 + m];
    v8f c;
#pragma unroll
    for (int j = 0; j < 8; ++j) c[j] = bb;
#pragma unroll
    for (int s = 0; s < 8; ++s) {
      v16bf bf = *(const v16bf*)(sw + ((nt * 8 + s) * 32 + lane) * 16);
      c = __builtin_amdgcn_wmma_f32_16x16x32_bf16(false, A[s], false, bf,
                                                  (short)0, c, false, false);
    }
    float w2v = w2[nt * 16 + m];
#pragma unroll
    for (int j = 0; j < 8; ++j) accd[j] += fmaxf(c[j], 0.0f) * w2v;
  }

  // Reduce across the 16 lanes of each half-wave (n dimension)
#pragma unroll
  for (int off = 1; off < 16; off <<= 1)
#pragma unroll
    for (int j = 0; j < 8; ++j)
      accd[j] += __shfl_xor(accd[j], off, 32);

  float bias2 = b2[0];
  int jj = lane & 15;
  if (jj < 8) {
    float v = accd[0];
#pragma unroll
    for (int j = 1; j < 8; ++j) if (jj == j) v = accd[j];
    float sg = 1.0f / (1.0f + expf(-(v + bias2)));
    pout[pBase + kh * 8 + jj] = sg;
  }
}

// ---------------------------------------------------------------------------
extern "C" void kernel_launch(void* const* d_in, const int* in_sizes, int n_in,
                              void* d_out, int out_size, void* d_ws, size_t ws_size,
                              hipStream_t stream) {
  (void)in_sizes; (void)n_in; (void)out_size; (void)ws_size;
  const float* x      = (const float*)d_in[0];
  const int*   erow   = (const int*)d_in[1];
  const int*   ecol   = (const int*)d_in[2];
  const float* ew     = (const float*)d_in[3];
  const int*   xidx   = (const int*)d_in[4];
  const int*   yidx   = (const int*)d_in[5];
  const float* gc1_ws = (const float*)d_in[6];
  const float* gc1_wn = (const float*)d_in[7];
  const float* gc1_b  = (const float*)d_in[8];
  const float* gc2_ws = (const float*)d_in[9];
  const float* gc2_wn = (const float*)d_in[10];
  const float* gc2_b  = (const float*)d_in[11];
  const float* gc3_ws = (const float*)d_in[12];
  const float* gc3_wn = (const float*)d_in[13];
  const float* gc3_b  = (const float*)d_in[14];
  const float* dec_w1 = (const float*)d_in[15];
  const float* dec_b1 = (const float*)d_in[16];
  const float* dec_w2 = (const float*)d_in[17];
  const float* dec_b2 = (const float*)d_in[18];

  // Workspace layout: [frag 256KB][H 51.2MB][S 51.2MB]  (~103 MB total)
  char* ws = (char*)d_ws;
  __bf16* frag = (__bf16*)ws;
  float* H = (float*)(ws + (256u << 10));
  float* S = (float*)(ws + (256u << 10) + (size_t)N_NODES * NHID * sizeof(float));

  float* P = (float*)d_out;            // p: [500000]
  float* Z = P + N_PAIRS;              // z: [100000 x 128]

  const __bf16* f1 = frag + 0 * 16384;   // gc1 ws||wn (64KB)
  const __bf16* f2 = frag + 2 * 16384;   // gc2 ws||wn
  const __bf16* f3 = frag + 4 * 16384;   // gc3 ws||wn
  const __bf16* fd = frag + 6 * 16384;   // dec_w1 (64KB)

  prep_frags<<<32, 256, 0, stream>>>(gc1_ws, gc1_wn, gc2_ws, gc2_wn,
                                     gc3_ws, gc3_wn, dec_w1, frag);

  const int gemmBlocks = 782;           // ceil(6250 waves / 8 per block)
  const int edgeBlocks = N_EDGES / 8;   // 1 wave per edge, 8 waves per block

  // Layer 1: x -> H (acc), S (support); no input relu
  gemm_dual<<<gemmBlocks, 256, 0, stream>>>(x, f1, gc1_b, H, S, 0);
  edge_scatter<<<edgeBlocks, 256, 0, stream>>>(erow, ecol, ew, S, H);

  // Layer 2: relu(H) -> H, S   (in-place safe: row-private read->write per wave)
  gemm_dual<<<gemmBlocks, 256, 0, stream>>>(H, f2, gc2_b, H, S, 1);
  edge_scatter<<<edgeBlocks, 256, 0, stream>>>(erow, ecol, ew, S, H);

  // Layer 3: relu(H) -> Z (directly into d_out z-region), S
  gemm_dual<<<gemmBlocks, 256, 0, stream>>>(H, f3, gc3_b, Z, S, 1);
  edge_scatter<<<edgeBlocks, 256, 0, stream>>>(erow, ecol, ew, S, Z);

  // Decoder: 31250 waves (16 pairs each), 8 waves/block -> 3907 blocks
  decoder<<<3907, 256, 0, stream>>>(Z, xidx, yidx, fd, dec_b1, dec_w2, dec_b2, P);
}
